// MSA_592705487565
// MI455X (gfx1250) — matrix operand order repaired and checked by reading, not compiled, not run on hardware
//
#include <hip/hip_runtime.h>
#include <math.h>

// ---------------------------------------------------------------------------
// MSA (LayerNorm + QKV + attention(64 heads x d=12, padded to 16) + out proj)
// bf16 WMMA pipeline for gfx1250 (wave32), with TDM + async-LDS staging in
// the two big GEMMs.
// ---------------------------------------------------------------------------

typedef __attribute__((ext_vector_type(16))) __bf16 v16bf;
typedef __attribute__((ext_vector_type(8)))  __bf16 v8bf;
typedef __attribute__((ext_vector_type(8)))  float  v8f;
typedef unsigned int v4u __attribute__((ext_vector_type(4)));
typedef int          v8i __attribute__((ext_vector_type(8)));
typedef int          v4i __attribute__((ext_vector_type(4)));

#define DIM      768
#define B        4
#define N        1024
#define EFF_H    64
#define EFF_D    12
#define DPAD     16
#define QKV_COLS 2304          // 3 * 768
#define TOKENS   4096          // B * N
#define SCALE    0.125f        // 64^-0.5
#define KCHUNK   64

// ---------------- workspace layout (bytes) ----------------
#define OFF_H      ((size_t)0)                                   // 4096*768 bf16
#define OFF_WQKVT  (OFF_H     + (size_t)TOKENS*DIM*2)            // 2304*768 bf16
#define OFF_WOUTT  (OFF_WQKVT + (size_t)QKV_COLS*DIM*2)          // 768*768 bf16
#define OFF_Q      (OFF_WOUTT + (size_t)DIM*DIM*2)               // b*h*n*16 bf16
#define OFF_K      (OFF_Q     + (size_t)B*EFF_H*N*DPAD*2)
#define OFF_VT     (OFF_K     + (size_t)B*EFF_H*N*DPAD*2)        // b*h*16*n bf16
#define OFF_WV     (OFF_VT    + (size_t)B*EFF_H*DPAD*N*2)        // 4096*768 bf16

// ---------------------------------------------------------------------------
// TDM: DMA a 64x64 bf16 tile (row-major, row stride 768 elements) into LDS.
// D# built per CDNA5 ISA ch.8: group0 {count=1, lds_addr, global_addr, type=2},
// group1 {data_size=2B, tensor_dim0=768, tensor_dim1=rows, tile 64x64,
// dim0_stride=768}, groups 2/3 zero (<=2D tensor).
// This toolchain's builtin takes 6 args (extra int32x8 group before cpol).
// ---------------------------------------------------------------------------
__device__ __forceinline__ void tdm_load_tile64x64(unsigned lds_off,
                                                   const __bf16* g,
                                                   unsigned tensor_rows) {
  unsigned long long ga = (unsigned long long)(size_t)g;
  v4u g0;
  g0[0] = 1u;                                   // count=1, user descriptor
  g0[1] = lds_off;                              // lds_addr
  g0[2] = (unsigned)ga;                         // global_addr[31:0]
  g0[3] = (unsigned)((ga >> 32) & 0x1FFFFFFu)   // global_addr[56:32]
          | 0x80000000u;                        // type=2 ("image")
  v8i g1;
  g1[0] = 0x00010000;                           // wg_mask=0, data_size=1 (2B)
  g1[1] = (int)(768u << 16);                    // tensor_dim0 = 768
  g1[2] = (int)((tensor_rows & 0xFFFFu) << 16); // tensor_dim1
  g1[3] = (int)(64u << 16);                     // tile_dim0 = 64
  g1[4] = 64;                                   // tile_dim1 = 64
  g1[5] = 768;                                  // tensor_dim0_stride = 768
  g1[6] = 0;
  g1[7] = 0;
  v4i gz = {0, 0, 0, 0};
  v8i gz8 = {0, 0, 0, 0, 0, 0, 0, 0};
  __builtin_amdgcn_tensor_load_to_lds(g0, g1, gz, gz, gz8, 0);
}

// async 16-byte global -> LDS copy (per-lane), tracked by ASYNCcnt
__device__ __forceinline__ void async_copy_b128(unsigned lds_off,
                                                const void* src) {
  unsigned long long ga = (unsigned long long)(size_t)src;
  asm volatile("global_load_async_to_lds_b128 %0, %1, off"
               :: "v"(lds_off), "v"(ga)
               : "memory");
}

__device__ __forceinline__ void wait_asynccnt0() {
  asm volatile("s_wait_asynccnt 0x0" ::: "memory");
}

// ---------------------------------------------------------------------------
// 1) cast + transpose weights to bf16
// ---------------------------------------------------------------------------
__global__ __launch_bounds__(256) void cast_weights_kernel(
    const float* __restrict__ w_qkv, const float* __restrict__ w_out,
    __bf16* __restrict__ wqkvT, __bf16* __restrict__ woutT) {
  const int t1 = QKV_COLS * DIM;
  const int t2 = DIM * DIM;
  int idx = blockIdx.x * 256 + threadIdx.x;
  if (idx < t1) {
    int n = idx / DIM, k = idx - n * DIM;
    wqkvT[idx] = (__bf16)w_qkv[(size_t)k * QKV_COLS + n];
  } else if (idx < t1 + t2) {
    int j = idx - t1;
    int n = j / DIM, k = j - n * DIM;
    woutT[j] = (__bf16)w_out[(size_t)k * DIM + n];
  }
}

// ---------------------------------------------------------------------------
// 2) zero padded q/k/vT buffers (contiguous region)
// ---------------------------------------------------------------------------
__global__ __launch_bounds__(256) void zero_kernel(unsigned int* __restrict__ p,
                                                   long nwords) {
  long i = (long)blockIdx.x * 256 + threadIdx.x;
  if (i < nwords) p[i] = 0u;
}

// ---------------------------------------------------------------------------
// 3) LayerNorm -> bf16 activations  (one block per token row)
// ---------------------------------------------------------------------------
__global__ __launch_bounds__(256) void ln_kernel(
    const float* __restrict__ x, const float* __restrict__ gamma,
    const float* __restrict__ beta, __bf16* __restrict__ hout) {
  __shared__ float red[256];
  const int row = blockIdx.x;
  const int t = threadIdx.x;
  const float* xr = x + (size_t)row * DIM;
  float v0 = xr[t], v1 = xr[t + 256], v2 = xr[t + 512];

  red[t] = v0 + v1 + v2;
  __syncthreads();
  for (int o = 128; o > 0; o >>= 1) {
    if (t < o) red[t] += red[t + o];
    __syncthreads();
  }
  const float mu = red[0] * (1.0f / DIM);
  __syncthreads();

  float d0 = v0 - mu, d1 = v1 - mu, d2 = v2 - mu;
  red[t] = d0 * d0 + d1 * d1 + d2 * d2;
  __syncthreads();
  for (int o = 128; o > 0; o >>= 1) {
    if (t < o) red[t] += red[t + o];
    __syncthreads();
  }
  const float rs = rsqrtf(red[0] * (1.0f / DIM) + 1e-5f);

  __bf16* ho = hout + (size_t)row * DIM;
  ho[t]       = (__bf16)(d0 * rs * gamma[t]       + beta[t]);
  ho[t + 256] = (__bf16)(d1 * rs * gamma[t + 256] + beta[t + 256]);
  ho[t + 512] = (__bf16)(d2 * rs * gamma[t + 512] + beta[t + 512]);
}

// ---------------------------------------------------------------------------
// WMMA from LDS tiles.
// A fragment (16x32 bf16): lane(0-15)->M, contiguous 16 K-values at 16*(lane>>4).
// B fragment (32x16 bf16): lane->N, contiguous 16 K-values at 16*(lane>>4).
// C: lane->N, vgpr r -> M = r + 8*(lane>>4).
// ---------------------------------------------------------------------------
__device__ __forceinline__ v8f wmma_lds(const __bf16* __restrict__ a_row,
                                        const __bf16* __restrict__ b_col,
                                        int ksub, int hl, v8f acc) {
  v8bf alo = *(const v8bf*)(a_row + ksub + hl * 16);
  v8bf ahi = *(const v8bf*)(a_row + ksub + hl * 16 + 8);
  v16bf a;
#pragma unroll
  for (int e = 0; e < 8; ++e) { a[e] = alo[e]; a[e + 8] = ahi[e]; }
  v16bf b = *(const v16bf*)(b_col + ksub + hl * 16);
  return __builtin_amdgcn_wmma_f32_16x16x32_bf16(false, a, false, b, (short)0,
                                                 acc, false, false);
}

// ---------------------------------------------------------------------------
// Shared GEMM core (M tile 64, N tile 32, K=768):
//  - wave 0 issues one TDM load for the 64x64 A tile (TENSORcnt)
//  - all 256 threads issue one async b128 for the 32x64 B tile (ASYNCcnt)
//  - 8 waves (4 along M x 2 along N) each run 2 WMMAs per K-chunk from LDS
// ---------------------------------------------------------------------------
__device__ __forceinline__ v8f gemm_tile_core(
    const __bf16* __restrict__ A, const __bf16* __restrict__ BT,
    int m_block, int n_block, unsigned a_rows,
    __bf16* lds_a /*64x64*/, __bf16* lds_b /*32x64*/) {
  const int w = threadIdx.x >> 5;
  const int wm = w & 3, wn = w >> 2;
  const int lane = threadIdx.x & 31;
  const int lrow = lane & 15;
  const int hl = lane >> 4;

  // per-thread B staging assignment: 32 cols x 64 K = 256 x (8 bf16)
  const int bcol = threadIdx.x >> 3;          // 0..31
  const int bkoff = (threadIdx.x & 7) * 8;    // 0,8,..,56
  const __bf16* bsrc = BT + (size_t)(n_block + bcol) * DIM + bkoff;
  const unsigned lds_b_off =
      (unsigned)(size_t)&lds_b[bcol * KCHUNK + bkoff];
  const unsigned lds_a_off = (unsigned)(size_t)&lds_a[0];

  const __bf16* a_row = &lds_a[(wm * 16 + lrow) * KCHUNK];
  const __bf16* b_col = &lds_b[(wn * 16 + lrow) * KCHUNK];

  v8f acc = {};
  for (int kk = 0; kk < DIM; kk += KCHUNK) {
    if ((threadIdx.x >> 5) == 0) {            // wave 0: TDM for the A tile
      tdm_load_tile64x64(lds_a_off, A + (size_t)m_block * DIM + kk, a_rows);
    }
    async_copy_b128(lds_b_off, bsrc + kk);    // everyone: 16B of the B tile
    if (kk + KCHUNK < DIM)                    // prefetch next B chunk
      __builtin_prefetch(bsrc + kk + KCHUNK, 0, 1);

    wait_asynccnt0();
    __builtin_amdgcn_s_wait_tensorcnt(0);
    __syncthreads();

    acc = wmma_lds(a_row, b_col, 0, hl, acc);
    acc = wmma_lds(a_row, b_col, 32, hl, acc);
    __syncthreads();
  }
  return acc;
}

// ---------------------------------------------------------------------------
// 4) QKV GEMM [4096 x 2304 x 768] with rearrange epilogue:
//    col -> (sel, head, d);  q *= SCALE;  v stored transposed [b,h,d,n].
// ---------------------------------------------------------------------------
__global__ __launch_bounds__(256) void gemm_qkv_kernel(
    const __bf16* __restrict__ h, const __bf16* __restrict__ wqkvT,
    __bf16* __restrict__ qbuf, __bf16* __restrict__ kbuf,
    __bf16* __restrict__ vtbuf) {
  __shared__ __align__(32) __bf16 lds_a[64 * KCHUNK];
  __shared__ __align__(32) __bf16 lds_b[32 * KCHUNK];

  const int w = threadIdx.x >> 5;
  const int wm = w & 3, wn = w >> 2;
  const int m0 = blockIdx.y * 64 + wm * 16;
  const int n0 = blockIdx.x * 32 + wn * 16;

  v8f acc = gemm_tile_core(h, wqkvT, blockIdx.y * 64, blockIdx.x * 32,
                           TOKENS, lds_a, lds_b);

  const int lane = threadIdx.x & 31;
  const int lrow = lane & 15, hl = lane >> 4;
  const int col = n0 + lrow;
  const int sel = col / DIM;
  const int rem = col - sel * DIM;
  const int head = rem / EFF_D;
  const int dd = rem - head * EFF_D;

#pragma unroll
  for (int r = 0; r < 8; ++r) {
    const int m = m0 + r + 8 * hl;        // global token index
    const int bb = m >> 10;
    const int nn = m & 1023;
    const float val = acc[r];
    const size_t qk_idx = ((size_t)(bb * EFF_H + head) * N + nn) * DPAD + dd;
    if (sel == 0)      qbuf[qk_idx] = (__bf16)(val * SCALE);
    else if (sel == 1) kbuf[qk_idx] = (__bf16)val;
    else vtbuf[((size_t)(bb * EFF_H + head) * DPAD + dd) * N + nn] = (__bf16)val;
  }
}

// ---------------------------------------------------------------------------
// 5) flash attention: one wave per (b, head, 16-row q tile).
//    32 keys per iteration: 2 score WMMAs + LDS transpose + 1 P*V WMMA.
// ---------------------------------------------------------------------------
__global__ __launch_bounds__(32) void attn_kernel(
    const __bf16* __restrict__ qbuf, const __bf16* __restrict__ kbuf,
    const __bf16* __restrict__ vtbuf, __bf16* __restrict__ wvbuf) {
  __shared__ __align__(32) __bf16 lds_p[16 * 32];

  const int gid = blockIdx.x;
  const int qt = gid & 63;
  const int hh = (gid >> 6) & 63;
  const int bb = gid >> 12;

  const int lane = threadIdx.x & 31;
  const int lrow = lane & 15;
  const int hl = lane >> 4;

  const __bf16* qp = qbuf + ((size_t)(bb * EFF_H + hh) * N + qt * 16) * DPAD;
  const __bf16* kp = kbuf + (size_t)(bb * EFF_H + hh) * N * DPAD;
  const __bf16* vp = vtbuf + (size_t)(bb * EFF_H + hh) * DPAD * N;

  // Q A-fragment: K-dim = d, zero-padded 16..31
  v16bf qa;
  {
    v16bf qfull = *(const v16bf*)(qp + lrow * DPAD);
#pragma unroll
    for (int e = 0; e < 8; ++e) { qa[e] = hl ? (__bf16)0.0f : qfull[e]; qa[e + 8] = hl ? (__bf16)0.0f : qfull[e + 8]; }
  }

  float mrow[8], lacc[8];
#pragma unroll
  for (int r = 0; r < 8; ++r) { mrow[r] = -1e30f; lacc[r] = 0.0f; }
  v8f O = {};
  v16bf zv = {};

  for (int k0 = 0; k0 < N; k0 += 32) {
    // K B-fragments (two 16-key tiles); lanes 16-31 hold K=16..31 -> zeros
    v16bf kb0 = hl ? zv : *(const v16bf*)(kp + (size_t)(k0 + lrow) * DPAD);
    v16bf kb1 = hl ? zv : *(const v16bf*)(kp + (size_t)(k0 + 16 + lrow) * DPAD);

    v8f zc = {};
    v8f s1 = __builtin_amdgcn_wmma_f32_16x16x32_bf16(false, qa, false, kb0,
                                                     (short)0, zc, false, false);
    v8f s2 = __builtin_amdgcn_wmma_f32_16x16x32_bf16(false, qa, false, kb1,
                                                     (short)0, zc, false, false);

#pragma unroll
    for (int r = 0; r < 8; ++r) {
      float tmax = fmaxf(s1[r], s2[r]);
      tmax = fmaxf(tmax, __shfl_xor(tmax, 1, 32));
      tmax = fmaxf(tmax, __shfl_xor(tmax, 2, 32));
      tmax = fmaxf(tmax, __shfl_xor(tmax, 4, 32));
      tmax = fmaxf(tmax, __shfl_xor(tmax, 8, 32));
      const float mnew = fmaxf(mrow[r], tmax);
      const float alpha = __expf(mrow[r] - mnew);
      const float p1 = __expf(s1[r] - mnew);
      const float p2 = __expf(s2[r] - mnew);
      float ps = p1 + p2;
      ps += __shfl_xor(ps, 1, 32);
      ps += __shfl_xor(ps, 2, 32);
      ps += __shfl_xor(ps, 4, 32);
      ps += __shfl_xor(ps, 8, 32);
      lacc[r] = lacc[r] * alpha + ps;
      mrow[r] = mnew;
      O[r] *= alpha;
      const int row = r + 8 * hl;
      lds_p[row * 32 + lrow]      = (__bf16)p1;
      lds_p[row * 32 + 16 + lrow] = (__bf16)p2;
    }
    __syncthreads();

    // P A-fragment from LDS (transpose C-layout -> A-layout)
    v16bf pa;
    {
      v8bf plo = *(const v8bf*)(&lds_p[lrow * 32 + hl * 16]);
      v8bf phi = *(const v8bf*)(&lds_p[lrow * 32 + hl * 16 + 8]);
#pragma unroll
      for (int e = 0; e < 8; ++e) { pa[e] = plo[e]; pa[e + 8] = phi[e]; }
    }
    // V B-fragment: N = d (lane), K = key offset, contiguous from vT
    v16bf vb = *(const v16bf*)(vp + (size_t)lrow * N + k0 + hl * 16);

    O = __builtin_amdgcn_wmma_f32_16x16x32_bf16(false, pa, false, vb,
                                                (short)0, O, false, false);
    __syncthreads();
  }

  // epilogue: divide by softmax denominator, store [b, n, h*12 + d]
  const int d = lrow;
  if (d < EFF_D) {
#pragma unroll
    for (int r = 0; r < 8; ++r) {
      const float inv = 1.0f / lacc[r];
      const int row = qt * 16 + r + 8 * hl;
      wvbuf[((size_t)bb * N + row) * DIM + hh * EFF_D + d] =
          (__bf16)(O[r] * inv);
    }
  }
}

// ---------------------------------------------------------------------------
// 6) output GEMM [4096 x 768 x 768] -> fp32 d_out
// ---------------------------------------------------------------------------
__global__ __launch_bounds__(256) void gemm_out_kernel(
    const __bf16* __restrict__ wvbuf, const __bf16* __restrict__ woutT,
    float* __restrict__ out) {
  __shared__ __align__(32) __bf16 lds_a[64 * KCHUNK];
  __shared__ __align__(32) __bf16 lds_b[32 * KCHUNK];

  const int w = threadIdx.x >> 5;
  const int wm = w & 3, wn = w >> 2;
  const int m0 = blockIdx.y * 64 + wm * 16;
  const int n0 = blockIdx.x * 32 + wn * 16;

  v8f acc = gemm_tile_core(wvbuf, woutT, blockIdx.y * 64, blockIdx.x * 32,
                           TOKENS, lds_a, lds_b);

  const int lane = threadIdx.x & 31;
  const int lrow = lane & 15, hl = lane >> 4;
  const int col = n0 + lrow;
#pragma unroll
  for (int r = 0; r < 8; ++r) {
    const int m = m0 + r + 8 * hl;
    out[(size_t)m * DIM + col] = acc[r];
  }
}

// ---------------------------------------------------------------------------
extern "C" void kernel_launch(void* const* d_in, const int* in_sizes, int n_in,
                              void* d_out, int out_size, void* d_ws,
                              size_t ws_size, hipStream_t stream) {
  const float* x        = (const float*)d_in[0];
  const float* ln_gamma = (const float*)d_in[1];
  const float* ln_beta  = (const float*)d_in[2];
  const float* w_qkv    = (const float*)d_in[3];
  const float* w_out    = (const float*)d_in[4];
  float* out = (float*)d_out;

  char* ws = (char*)d_ws;
  __bf16* hbuf   = (__bf16*)(ws + OFF_H);
  __bf16* wqkvT  = (__bf16*)(ws + OFF_WQKVT);
  __bf16* woutT  = (__bf16*)(ws + OFF_WOUTT);
  __bf16* qbuf   = (__bf16*)(ws + OFF_Q);
  __bf16* kbuf   = (__bf16*)(ws + OFF_K);
  __bf16* vtbuf  = (__bf16*)(ws + OFF_VT);
  __bf16* wvbuf  = (__bf16*)(ws + OFF_WV);

  // 1) cast/transpose weights
  {
    const int total = QKV_COLS * DIM + DIM * DIM;
    cast_weights_kernel<<<(total + 255) / 256, 256, 0, stream>>>(
        w_qkv, w_out, wqkvT, woutT);
  }
  // 2) zero padded q/k/vT region (contiguous)
  {
    const long nwords = (long)(3 * (size_t)B * EFF_H * N * DPAD * 2) / 4;
    zero_kernel<<<(unsigned)((nwords + 255) / 256), 256, 0, stream>>>(
        (unsigned int*)qbuf, nwords);
  }
  // 3) layernorm
  ln_kernel<<<TOKENS, 256, 0, stream>>>(x, ln_gamma, ln_beta, hbuf);
  // 4) qkv gemm + rearrange
  gemm_qkv_kernel<<<dim3(QKV_COLS / 32, TOKENS / 64), 256, 0, stream>>>(
      hbuf, wqkvT, qbuf, kbuf, vtbuf);
  // 5) flash attention
  attn_kernel<<<B * EFF_H * (N / 16), 32, 0, stream>>>(qbuf, kbuf, vtbuf,
                                                       wvbuf);
  // 6) output projection
  gemm_out_kernel<<<dim3(DIM / 32, TOKENS / 64), 256, 0, stream>>>(
      wvbuf, woutT, out);
}
